// DSAM_21217138442424
// MI455X (gfx1250) — compile-verified, matrix operand order checked by reading, not run on hardware
//
#include <hip/hip_runtime.h>
#include <hip/hip_bf16.h>

// ---- problem constants (from reference) ----
#define BATCH   8
#define GROUPS_ 4
#define DH      64      // dim_head / per-group channels
#define NP      1024    // H*W query positions
#define NJ      64      // h2*w2 kv positions
#define SCALE_  0.125f  // 64^-0.5
#define OFFSCALE 4.0f

typedef __attribute__((ext_vector_type(2)))  float    v2f;
typedef __attribute__((ext_vector_type(8)))  float    v8f;
typedef __attribute__((ext_vector_type(16))) _Float16 v16h;

__device__ __forceinline__ v8f wmma4(v2f a, v2f b, v8f c) {
    // D = A(16x4 f32) * B(4x16 f32) + C(16x16 f32)
    return __builtin_amdgcn_wmma_f32_16x16x4_f32(
        false, a, false, b, (short)0, c, false, false);
}

__device__ __forceinline__ v8f wmma16h(v16h a, v16h b, v8f c) {
    // D = A(16x32 f16) * B(32x16 f16) + C(16x16 f32)
    return __builtin_amdgcn_wmma_f32_16x16x32_f16(
        false, a, false, b, (short)0, c, false, false);
}

// Load a 16-bit A/B WMMA fragment from a row-major LDS row.
// Per ISA 7.12.2 the lane's 16 halves are K in [off, off+7] and [off+16, off+23]
// (off = 8*hi + 32*ks) -> two contiguous 16B runs -> 2x ds_read_b128.
__device__ __forceinline__ v16h load_frag_h(const _Float16* rowbase, int off) {
    union { v16h h; uint4 q[2]; } u;
    u.q[0] = *(const uint4*)(rowbase + off);
    u.q[1] = *(const uint4*)(rowbase + off + 16);
    return u.h;
}

// ---------------------------------------------------------------------------
// Generic f32 WMMA GEMM:  Out[b][Co][P] = W[b%wmod][Co][Ci] @ X[b][Ci][P] (+bias)
// 4 waves per 128-thread block, one 16x16 output tile per wave.
// ---------------------------------------------------------------------------
__global__ __launch_bounds__(128)
void gemm_wmma_kernel(const float* __restrict__ W, const float* __restrict__ X,
                      float* __restrict__ Out, const float* __restrict__ bias,
                      int Ci, int Co, int P, int wmod)
{
    const int batch = blockIdx.x;
    const int wave  = threadIdx.x >> 5;
    const int lane  = threadIdx.x & 31;
    const int l16   = lane & 15;
    const int hi2   = (lane >> 4) << 1;      // 0 or 2
    const int ptiles = P >> 4;
    const int ntiles = (Co >> 4) * ptiles;
    const int tile = blockIdx.y * 4 + wave;  // wave-uniform
    if (tile >= ntiles) return;
    const int ot = tile / ptiles;
    const int pt = tile % ptiles;

    const float* Wb = W + (size_t)(batch % wmod) * Co * Ci;
    const float* Xb = X + (size_t)batch * Ci * P;
    float*       Ob = Out + (size_t)batch * Co * P;

    const int arow = ot * 16 + l16;
    const int col  = pt * 16 + l16;

    v8f acc = {0.f,0.f,0.f,0.f,0.f,0.f,0.f,0.f};
    for (int kk = 0; kk < Ci; kk += 4) {
        const int kb = kk + hi2;
        v2f a, b;
        a.x = Wb[arow * Ci + kb];
        a.y = Wb[arow * Ci + kb + 1];
        b.x = Xb[kb * P + col];
        b.y = Xb[(kb + 1) * P + col];
        acc = wmma4(a, b, acc);
    }
#pragma unroll
    for (int r = 0; r < 8; ++r) {
        const int orow = ot * 16 + r + (hi2 << 2);
        float v = acc[r];
        if (bias) v += bias[orow];
        Ob[orow * P + col] = v;
    }
}

// ---------------------------------------------------------------------------
// Offsets: depthwise 6x6 s4 conv on grouped q + bias + exact GELU + 1x1(64->2)
// + tanh*4, add base grid, normalize -> vgs[bg][j][2]
// ---------------------------------------------------------------------------
__global__ __launch_bounds__(64)
void offsets_kernel(const float* __restrict__ q_buf,
                    const float* __restrict__ wdw, const float* __restrict__ bdw,
                    const float* __restrict__ wpw, float* __restrict__ vgs)
{
    const int bg  = blockIdx.x;
    const int tid = threadIdx.x;          // 0..63 -> (oh,ow)
    const int oh = tid >> 3, ow = tid & 7;
    const float* qb = q_buf + (size_t)bg * DH * NP;

    float s0 = 0.f, s1 = 0.f;
    for (int c = 0; c < DH; ++c) {
        float acc = bdw[c];
        const float* qc = qb + c * NP;
        const float* wc = wdw + c * 36;
#pragma unroll
        for (int ky = 0; ky < 6; ++ky) {
            const int iy = oh * 4 - 1 + ky;
            if (iy < 0 || iy > 31) continue;
#pragma unroll
            for (int kx = 0; kx < 6; ++kx) {
                const int ix = ow * 4 - 1 + kx;
                if (ix < 0 || ix > 31) continue;
                acc = fmaf(qc[iy * 32 + ix], wc[ky * 6 + kx], acc);
            }
        }
        const float g = 0.5f * acc * (1.f + erff(acc * 0.70710678118f));
        s0 = fmaf(g, wpw[c], s0);
        s1 = fmaf(g, wpw[64 + c], s1);
    }
    const float vg0 = (float)ow + tanhf(s0) * OFFSCALE;  // x channel
    const float vg1 = (float)oh + tanhf(s1) * OFFSCALE;  // y channel
    vgs[bg * 128 + tid * 2 + 0] = 2.f * vg0 / 7.f - 1.f;
    vgs[bg * 128 + tid * 2 + 1] = 2.f * vg1 / 7.f - 1.f;
}

// ---------------------------------------------------------------------------
// Bilinear grid_sample (zeros, align_corners=False): kv[bg][c][j]
// ---------------------------------------------------------------------------
__global__ __launch_bounds__(64)
void sample_kernel(const float* __restrict__ x, const float* __restrict__ vgs,
                   float* __restrict__ kv)
{
    const int bg = blockIdx.x;
    const int j  = blockIdx.y;
    const int c  = threadIdx.x;
    const float gx = vgs[bg * 128 + j * 2 + 0];
    const float gy = vgs[bg * 128 + j * 2 + 1];
    const float ixf = ((gx + 1.f) * 32.f - 1.f) * 0.5f;
    const float iyf = ((gy + 1.f) * 32.f - 1.f) * 0.5f;
    const float x0f = floorf(ixf), y0f = floorf(iyf);
    const int x0 = (int)x0f, y0 = (int)y0f;
    const float wa = (x0f + 1.f - ixf) * (y0f + 1.f - iyf);
    const float wb = (ixf - x0f) * (y0f + 1.f - iyf);
    const float wc = (x0f + 1.f - ixf) * (iyf - y0f);
    const float wd = (ixf - x0f) * (iyf - y0f);
    const int b = bg >> 2, g = bg & 3;
    const float* img = x + (size_t)b * 262144 + (size_t)(g * DH + c) * NP;
    auto fetch = [&](int yy, int xx) -> float {
        if (xx < 0 || xx > 31 || yy < 0 || yy > 31) return 0.f;
        return img[yy * 32 + xx];
    };
    const float v = fetch(y0, x0) * wa + fetch(y0, x0 + 1) * wb +
                    fetch(y0 + 1, x0) * wc + fetch(y0 + 1, x0 + 1) * wd;
    kv[(size_t)bg * DH * NJ + c * NJ + j] = v;
}

// ---------------------------------------------------------------------------
// Fused attention: per (bg, 16-query tile):
//   sim = SCALE*Q@K^T + b3 (f32 WMMA)
//   CPB MLP bias: layer1 VALU -> f16 LDS; layer2 = f16 WMMA 16x16x32 with
//   register-resident B fragments and b128 LDS fragment loads; layer3 folded
//   per-lane, ds_add_f32 into sim
//   softmax ; out = attn@V (f32 WMMA) -> O[b][256][1024]
// ---------------------------------------------------------------------------
__global__ __launch_bounds__(128)
void attn_kernel(const float* __restrict__ q_buf, const float* __restrict__ k_buf,
                 const float* __restrict__ v_buf, const float* __restrict__ vgs,
                 const float* __restrict__ w1, const float* __restrict__ b1,
                 const float* __restrict__ w2, const float* __restrict__ b2,
                 const float* __restrict__ w3, const float* __restrict__ b3,
                 float* __restrict__ obuf)
{
    __shared__ float    sim_s[16 * 64];
    __shared__ _Float16 hid1h[4][16 * 64];   // per-wave f16 activations
    __shared__ _Float16 w2h[64 * 64];        // f16 copy of cpb_w2
    __shared__ float    featx[64 * 16];
    __shared__ float    featy[64 * 16];
    __shared__ float    vgs_j[128];
    __shared__ float    w1s[128];
    __shared__ float    b1s[64];
    __shared__ float    qxn[16], qyn[16];
    __shared__ float    red[16 * 8];
    __shared__ float    rowmax_s[16];
    __shared__ float    rinv_s[16];

    const int bg = blockIdx.x;          // b*4+g, g == head
    const int it = blockIdx.y;          // query tile
    const int i0 = it * 16;
    const int b  = bg >> 2;
    const int g  = bg & 3;
    const int tid  = threadIdx.x;
    const int wave = tid >> 5;
    const int lane = tid & 31;
    const int l16  = lane & 15;
    const int hi   = lane >> 4;         // 0 or 1
    const int hi2  = hi << 1;           // 0 or 2
    const int hi8  = hi << 3;           // 0 or 8
    const int mbase = hi << 3;          // 0 or 8 (C/D row base)

    // pull the shared CPB tables through cache early (global_prefetch_b8)
    __builtin_prefetch(w2 + tid * 32, 0, 1);

    vgs_j[tid] = vgs[bg * 128 + tid];   // 128 floats by 128 threads
    w1s[tid]   = w1[tid];               // cpb_w1 is 64x2 = 128 floats
    if (tid < 64) b1s[tid] = b1[tid];
    if (tid < 16) {
        const int i = i0 + tid;
        qxn[tid] = 2.f * (float)(i & 31) / 31.f - 1.f;
        qyn[tid] = 2.f * (float)(i >> 5) / 31.f - 1.f;
    }
    __syncthreads();

    // sign*log1p features feat[j][m] + stage w2 as f16 in LDS
    for (int e = tid; e < 1024; e += 128) {
        const int j = e >> 4, m = e & 15;
        const float fx = qxn[m] - vgs_j[j * 2 + 0];
        const float fy = qyn[m] - vgs_j[j * 2 + 1];
        featx[e] = copysignf(log1pf(fabsf(fx)), fx);
        featy[e] = copysignf(log1pf(fabsf(fy)), fy);
    }
    for (int e = tid; e < 4096; e += 128) w2h[e] = (_Float16)w2[e];

    // sim tile: wave w covers j in [16w, 16w+16)  (f32 WMMA, K=64)
    {
        const float* qb = q_buf + (size_t)bg * DH * NP;
        const float* kb = k_buf + (size_t)bg * DH * NJ;  // [d][j] == B[k][n]
        const int n0 = wave * 16;
        v8f acc = {0.f,0.f,0.f,0.f,0.f,0.f,0.f,0.f};
        for (int kk = 0; kk < DH; kk += 4) {
            const int kb2 = kk + hi2;
            v2f a, bb;
            a.x = qb[kb2 * NP + i0 + l16];
            a.y = qb[(kb2 + 1) * NP + i0 + l16];
            bb.x = kb[kb2 * NJ + n0 + l16];
            bb.y = kb[(kb2 + 1) * NJ + n0 + l16];
            acc = wmma4(a, bb, acc);
        }
        const float b3v = b3[0];
#pragma unroll
        for (int r = 0; r < 8; ++r)
            sim_s[(mbase + r) * 64 + n0 + l16] = acc[r] * SCALE_ + b3v;
    }
    __syncthreads();

    // ---- CPB bias: each wave handles 16 j-columns ----
    // Preload ALL layer-2 B fragments (w2, invariant over jt) into registers:
    // 4 n-tiles x 2 K-steps x 8 VGPRs = 64 VGPRs.
    v16h ubf[4][2];
    float b2o[4], w3o[4];
#pragma unroll
    for (int nt = 0; nt < 4; ++nt) {
        const int o = nt * 16 + l16;
#pragma unroll
        for (int ks = 0; ks < 2; ++ks)
            ubf[nt][ks] = load_frag_h(&w2h[o * 64], hi8 + (ks << 5));
        b2o[nt] = b2[o];
        w3o[nt] = w3[o];
    }

    _Float16* h1 = hid1h[wave];
    for (int jt = wave; jt < 64; jt += 4) {
        for (int t = 0; t < 32; ++t) {          // hid1: 1024 entries / 32 lanes
            const int e = lane + (t << 5);
            const int m = e >> 6, cc = e & 63;
            const float h = fmaf(w1s[cc * 2 + 0], featx[jt * 16 + m],
                            fmaf(w1s[cc * 2 + 1], featy[jt * 16 + m], b1s[cc]));
            h1[m * 64 + cc] = (_Float16)fmaxf(h, 0.f);
        }
        // A fragments for this jt (2 K-steps), 2x ds_read_b128 each
        const v16h ua0 = load_frag_h(&h1[l16 * 64], hi8);
        const v16h ua1 = load_frag_h(&h1[l16 * 64], hi8 + 32);

        float part[8] = {0.f,0.f,0.f,0.f,0.f,0.f,0.f,0.f};
#pragma unroll
        for (int nt = 0; nt < 4; ++nt) {
            v8f acc = {0.f,0.f,0.f,0.f,0.f,0.f,0.f,0.f};
            acc = wmma16h(ua0, ubf[nt][0], acc);
            acc = wmma16h(ua1, ubf[nt][1], acc);
#pragma unroll
            for (int r = 0; r < 8; ++r)
                part[r] = fmaf(fmaxf(acc[r] + b2o[nt], 0.f), w3o[nt], part[r]);
        }
#pragma unroll
        for (int r = 0; r < 8; ++r)             // reduce 16 lanes -> sim (ds_add_f32)
            atomicAdd(&sim_s[(mbase + r) * 64 + jt], part[r]);
    }
    __syncthreads();

    // softmax over 64 kv per query row
    {
        const int m = tid >> 3, seg = tid & 7;
        float mx = -1e30f;
        for (int c = seg * 8; c < seg * 8 + 8; ++c) mx = fmaxf(mx, sim_s[m * 64 + c]);
        red[m * 8 + seg] = mx;
        __syncthreads();
        if (seg == 0) {
            float r2 = red[m * 8];
            for (int s2 = 1; s2 < 8; ++s2) r2 = fmaxf(r2, red[m * 8 + s2]);
            rowmax_s[m] = r2;
        }
        __syncthreads();
        const float rm = rowmax_s[m];
        float s = 0.f;
        for (int c = seg * 8; c < seg * 8 + 8; ++c) {
            const float e = __expf(sim_s[m * 64 + c] - rm);
            sim_s[m * 64 + c] = e;
            s += e;
        }
        red[m * 8 + seg] = s;
        __syncthreads();
        if (seg == 0) {
            float r2 = 0.f;
            for (int s2 = 0; s2 < 8; ++s2) r2 += red[m * 8 + s2];
            rinv_s[m] = 1.f / r2;
        }
        __syncthreads();
    }

    // out = attn @ V : wave w covers d in [16w, 16w+16)  (f32 WMMA)
    {
        const float* vb = v_buf + (size_t)bg * DH * NJ;  // [d][j]; B[k=j][n=d]=vb[n*64+k]
        const int n0 = wave * 16;
        const float ri = rinv_s[l16];
        v8f acc = {0.f,0.f,0.f,0.f,0.f,0.f,0.f,0.f};
        for (int kk = 0; kk < NJ; kk += 4) {
            const int kb2 = kk + hi2;
            v2f a, bb;
            a.x = sim_s[l16 * 64 + kb2] * ri;
            a.y = sim_s[l16 * 64 + kb2 + 1] * ri;
            const float* vrow = vb + (n0 + l16) * 64 + kb2;
            bb.x = vrow[0]; bb.y = vrow[1];
            acc = wmma4(a, bb, acc);
        }
#pragma unroll
        for (int r = 0; r < 8; ++r) {
            const int m = mbase + r;
            const int d = n0 + l16;
            obuf[(size_t)b * 262144 + (size_t)(g * DH + d) * NP + (i0 + m)] = acc[r];
        }
    }
}

// ---------------------------------------------------------------------------
extern "C" void kernel_launch(void* const* d_in, const int* in_sizes, int n_in,
                              void* d_out, int out_size, void* d_ws, size_t ws_size,
                              hipStream_t stream) {
    const float* x        = (const float*)d_in[0];
    const float* wq       = (const float*)d_in[1];
    const float* wk       = (const float*)d_in[2];
    const float* wv       = (const float*)d_in[3];
    const float* w_off_dw = (const float*)d_in[4];
    const float* b_off_dw = (const float*)d_in[5];
    const float* w_off_pw = (const float*)d_in[6];
    const float* cpb_w1   = (const float*)d_in[7];
    const float* cpb_b1   = (const float*)d_in[8];
    const float* cpb_w2   = (const float*)d_in[9];
    const float* cpb_b2   = (const float*)d_in[10];
    const float* cpb_w3   = (const float*)d_in[11];
    const float* cpb_b3   = (const float*)d_in[12];
    const float* w_out    = (const float*)d_in[13];
    const float* b_out    = (const float*)d_in[14];
    float* out = (float*)d_out;

    // workspace carving (floats): ~18.4 MB total
    float* ws     = (float*)d_ws;
    float* q_buf  = ws;                  // 32*64*1024 = 2097152
    float* O_buf  = q_buf + 2097152;     // 8*256*1024 = 2097152
    float* kv_buf = O_buf + 2097152;     // 32*64*64   = 131072
    float* k_buf  = kv_buf + 131072;     // 131072
    float* v_buf  = k_buf  + 131072;     // 131072
    float* vgs    = v_buf  + 131072;     // 32*64*2    = 4096

    // 1) q = grouped 1x1 conv (WMMA): per bg 64x1024 = W[g]@X
    gemm_wmma_kernel<<<dim3(32, 64), 128, 0, stream>>>(wq, x, q_buf, nullptr, 64, 64, 1024, 4);
    // 2) offsets -> normalized sample grid vgs
    offsets_kernel<<<32, 64, 0, stream>>>(q_buf, w_off_dw, b_off_dw, w_off_pw, vgs);
    // 3) bilinear grid-sample -> kv[bg][c][j]
    sample_kernel<<<dim3(32, 64), 64, 0, stream>>>(x, vgs, kv_buf);
    // 4) k, v grouped 1x1 convs (WMMA)
    gemm_wmma_kernel<<<dim3(32, 4), 128, 0, stream>>>(wk, kv_buf, k_buf, nullptr, 64, 64, 64, 4);
    gemm_wmma_kernel<<<dim3(32, 4), 128, 0, stream>>>(wv, kv_buf, v_buf, nullptr, 64, 64, 64, 4);
    // 5) fused bias-MLP + attention (f16 WMMA hot loop) -> O[b][256][1024]
    attn_kernel<<<dim3(32, 64), 128, 0, stream>>>(q_buf, k_buf, v_buf, vgs,
        cpb_w1, cpb_b1, cpb_w2, cpb_b2, cpb_w3, cpb_b3, O_buf);
    // 6) final 1x1 conv 256->256 + bias (WMMA)
    gemm_wmma_kernel<<<dim3(8, 256), 128, 0, stream>>>(w_out, O_buf, out, b_out, 256, 256, 1024, 1);
}
